// Damping_28106265985274
// MI455X (gfx1250) — matrix operand order repaired and checked by reading, not compiled, not run on hardware
//
#include <hip/hip_runtime.h>

typedef _Float16 half_t;
typedef __attribute__((ext_vector_type(16))) _Float16 v16h;
typedef __attribute__((ext_vector_type(8)))  _Float16 v8h;
typedef __attribute__((ext_vector_type(8)))  float    v8f;

// Problem sizes (fixed by reference)
#define NN   64
#define HH   256
#define OFFN 2016   // 126 tiles of 16

// Workspace layout in half elements (packed f16 weights, WMMA-B fragment order)
static constexpr size_t WS_WD1 = 0;                         // 64x256
static constexpr size_t WS_WD2 = WS_WD1 + 64u * 256u;       // 256x256
static constexpr size_t WS_WDO = WS_WD2 + 256u * 256u;      // 256x64
static constexpr size_t WS_WO1 = WS_WDO + 256u * 64u;       // 64x256
static constexpr size_t WS_WO2 = WS_WO1 + 64u * 256u;       // 256x256
static constexpr size_t WS_WOO = WS_WO2 + 256u * 256u;      // 256x2016

// Branchless tanh from hardware transcendentals:
// tanh(x) = (e^{2x}-1)/(e^{2x}+1), e^{2x} = exp2(2*log2(e)*x), clamp |x|<=10
__device__ __forceinline__ float fast_tanh(float v) {
    float c = fminf(fmaxf(v, -10.0f), 10.0f);
    float e = __builtin_amdgcn_exp2f(c * 2.8853900817779268f); // v_exp_f32
    return (e - 1.0f) * __builtin_amdgcn_rcpf(e + 1.0f);       // v_rcp_f32
}

// ---------------------------------------------------------------------------
// Pack a row-major f32 weight W[K x N] into f16 blocks of 512 elements.
// Block index = nt * (K/32) + kt  (K-tiles contiguous for the K-loop stream).
// Within a block, element (lane*16 + e) holds W[k, n] with
//   n = nt*16 + (lane & 15)
//   k = kt*32 + 8*(lane>>4) + (e<8 ? e : e+8)   (B-matrix 32x16 f16 layout,
//                                      same k order as the A fragment)
// ---------------------------------------------------------------------------
__global__ void pack_weight_f16(const float* __restrict__ W,
                                half_t* __restrict__ dst, int K, int N) {
    int idx = blockIdx.x * blockDim.x + threadIdx.x;
    if (idx >= K * N) return;
    int q      = idx & 511;
    int blk    = idx >> 9;
    int ktiles = K >> 5;
    int kt     = blk % ktiles;
    int nt     = blk / ktiles;
    int lane   = q >> 4;
    int e      = q & 15;
    int n = nt * 16 + (lane & 15);
    int k = kt * 32 + ((lane >> 4) << 3) + (e < 8 ? e : e + 8);
    dst[idx] = (half_t)W[(size_t)k * N + n];
}

// ---------------------------------------------------------------------------
// A fragment: 16-bit A 16x32 layout. Lane holds row m = lane&15.
// k(e) = kbase + (e<8 ? e : e+8) + 8*(lane>>4)  -> two contiguous 16B chunks.
// ---------------------------------------------------------------------------
__device__ __forceinline__ v16h load_afrag(const half_t* __restrict__ act,
                                           int stride, int kbase, int lane) {
    int m  = lane & 15;
    int hh = lane >> 4;
    const half_t* row = act + m * stride + kbase + hh * 8;
    v8h lo = *(const v8h*)(row);        // k = kbase + 8*hh + 0..7
    v8h hi = *(const v8h*)(row + 16);   // k = kbase + 16 + 8*hh + 0..7
    v16h r;
#pragma unroll
    for (int e = 0; e < 8; ++e) { r[e] = lo[e]; r[8 + e] = hi[e]; }
    return r;
}

// One 16x16 output tile of [16 x K] @ [K x N], K multiple of 32.
__device__ __forceinline__ v8f gemm_tile(const half_t* __restrict__ actA, int strideA,
                                         const half_t* __restrict__ wPacked,
                                         int K, int nt, int lane) {
    v8f acc = {};
    int ktiles = K >> 5;
    const half_t* bblk = wPacked + (size_t)nt * ktiles * 512 + lane * 16;
#pragma unroll
    for (int kt = 0; kt < 8; ++kt) {          // max ktiles == 8
        if (kt >= ktiles) break;
        v16h a = load_afrag(actA, strideA, kt * 32, lane);
        v16h b = *(const v16h*)(bblk + (size_t)kt * 512);
        acc = __builtin_amdgcn_wmma_f32_16x16x32_f16(
                false, a, false, b, (short)0, acc, false, false);
    }
    return acc;
}

// bias + tanh epilogue -> f16 activation tile in LDS (row-major)
__device__ __forceinline__ void store_tanh(v8f acc, const float* __restrict__ bias,
                                           int nt, half_t* __restrict__ dstAct,
                                           int stride, int lane) {
    int n  = lane & 15;
    int hh = lane >> 4;
    float bv = bias[nt * 16 + n];
#pragma unroll
    for (int r = 0; r < 8; ++r) {
        int m = r + 8 * hh;
        dstAct[m * stride + nt * 16 + n] = (half_t)fast_tanh(acc[r] + bv);
    }
}

// ---------------------------------------------------------------------------
// Main fused kernel: one workgroup (8 waves / 256 threads) per 16 batch rows.
// ---------------------------------------------------------------------------
__global__ void __launch_bounds__(256)
damping_fused_kernel(const float* __restrict__ x,
                     const float* __restrict__ bd1, const float* __restrict__ bd2,
                     const float* __restrict__ bdo,
                     const float* __restrict__ bo1, const float* __restrict__ bo2,
                     const float* __restrict__ boo,
                     const float* __restrict__ damp_min,
                     const half_t* __restrict__ ws,
                     float* __restrict__ out) {
    extern __shared__ char smem_raw[];
    float*  sXf   = (float*)smem_raw;            // 16*64  f32
    float*  sDiag = sXf   + 16 * NN;             // 16*64  f32
    float*  sV    = sDiag + 16 * NN;             // 16*64  f32
    float*  sOff  = sV    + 16 * NN;             // 16*2016 f32
    half_t* sXh   = (half_t*)(sOff + 16 * OFFN); // 16*64  f16
    half_t* sActA = sXh   + 16 * NN;             // 16*256 f16
    half_t* sActB = sActA + 16 * HH;             // 16*256 f16

    const int tid     = threadIdx.x;
    const int lane    = tid & 31;
    const int wave    = tid >> 5;
    const int rowBase = blockIdx.x * 16;

    const half_t* wWd1 = ws + WS_WD1;
    const half_t* wWd2 = ws + WS_WD2;
    const half_t* wWdo = ws + WS_WDO;
    const half_t* wWo1 = ws + WS_WO1;
    const half_t* wWo2 = ws + WS_WO2;
    const half_t* wWoo = ws + WS_WOO;

    // ---- stage 0: load x tile (f32 + f16 copies) ----
    for (int t = tid; t < 16 * NN; t += 256) {
        int b = t >> 6, c = t & 63;
        float xv = x[(size_t)(rowBase + b) * NN + c];
        sXf[t] = xv;
        sXh[b * NN + c] = (half_t)xv;
    }
    __syncthreads();

    // ---- stage 1: h1 = tanh(x @ Wd1 + bd1) ----  (16 N-tiles, 2 per wave)
    for (int nt = wave; nt < HH / 16; nt += 8) {
        v8f acc = gemm_tile(sXh, NN, wWd1, NN, nt, lane);
        store_tanh(acc, bd1, nt, sActA, HH, lane);
    }
    __syncthreads();

    // ---- stage 2: h2 = tanh(h1 @ Wd2 + bd2) ----
    for (int nt = wave; nt < HH / 16; nt += 8) {
        v8f acc = gemm_tile(sActA, HH, wWd2, HH, nt, lane);
        store_tanh(acc, bd2, nt, sActB, HH, lane);
    }
    __syncthreads();

    // ---- stage 3: d = h2 @ Wdo + bdo ; diag = (relu(d)+damp_min)*x ----
    if (wave < 4) {
        int nt = wave;
        v8f acc = gemm_tile(sActB, HH, wWdo, HH, nt, lane);
        int n = lane & 15, hh = lane >> 4;
        float bv = bdo[nt * 16 + n];
        float dm = damp_min[nt * 16 + n];
#pragma unroll
        for (int r = 0; r < 8; ++r) {
            int m = r + 8 * hh;
            float dv = acc[r] + bv;
            dv = fmaxf(dv, 0.0f) + dm;
            sDiag[m * NN + nt * 16 + n] = dv * sXf[m * NN + nt * 16 + n];
        }
    }
    // ---- stage 3b: g1 = tanh(x @ Wo1 + bo1) (sActA is free again) ----
    for (int nt = wave; nt < HH / 16; nt += 8) {
        v8f acc = gemm_tile(sXh, NN, wWo1, NN, nt, lane);
        store_tanh(acc, bo1, nt, sActA, HH, lane);
    }
    __syncthreads();

    // ---- stage 4: g2 = tanh(g1 @ Wo2 + bo2) ----
    for (int nt = wave; nt < HH / 16; nt += 8) {
        v8f acc = gemm_tile(sActA, HH, wWo2, HH, nt, lane);
        store_tanh(acc, bo2, nt, sActB, HH, lane);
    }
    __syncthreads();

    // ---- stage 5: off = g2 @ Woo + boo -> LDS (126 N-tiles round-robin) ----
    for (int nt = wave; nt < OFFN / 16; nt += 8) {
        v8f acc = gemm_tile(sActB, HH, wWoo, HH, nt, lane);
        int n = lane & 15, hh = lane >> 4;
        float bv = boo[nt * 16 + n];
#pragma unroll
        for (int r = 0; r < 8; ++r) {
            int m = r + 8 * hh;
            sOff[m * OFFN + nt * 16 + n] = acc[r] + bv;
        }
    }
    __syncthreads();

    // ---- stage 6: v = L^T x  (v_j = diag_j*x_j + sum_{i>j} off[i(i-1)/2+j]*x_i) ----
    for (int t = tid; t < 16 * NN; t += 256) {
        int b = t >> 6, j = t & 63;
        const float* offr = sOff + b * OFFN;
        const float* xr   = sXf + b * NN;
        float acc = sDiag[b * NN + j] * xr[j];
        for (int i = j + 1; i < NN; ++i)
            acc += offr[(i * (i - 1)) / 2 + j] * xr[i];
        sV[b * NN + j] = acc;
    }
    __syncthreads();

    // ---- stage 7: out = L v  (out_i = diag_i*v_i + sum_{j<i} off_row[j]*v_j) ----
    for (int t = tid; t < 16 * NN; t += 256) {
        int b = t >> 6, i = t & 63;
        const float* offr = sOff + b * OFFN + (i * (i - 1)) / 2;
        const float* vr   = sV + b * NN;
        float acc = sDiag[b * NN + i] * vr[i];
        for (int j = 0; j < i; ++j)
            acc += offr[j] * vr[j];
        out[(size_t)(rowBase + b) * NN + i] = acc;
    }
}

extern "C" void kernel_launch(void* const* d_in, const int* in_sizes, int n_in,
                              void* d_out, int out_size, void* d_ws, size_t ws_size,
                              hipStream_t stream) {
    const float* x        = (const float*)d_in[0];
    const float* Wd1      = (const float*)d_in[1];
    const float* bd1      = (const float*)d_in[2];
    const float* Wd2      = (const float*)d_in[3];
    const float* bd2      = (const float*)d_in[4];
    const float* Wdo      = (const float*)d_in[5];
    const float* bdo      = (const float*)d_in[6];
    const float* Wo1      = (const float*)d_in[7];
    const float* bo1      = (const float*)d_in[8];
    const float* Wo2      = (const float*)d_in[9];
    const float* bo2      = (const float*)d_in[10];
    const float* Woo      = (const float*)d_in[11];
    const float* boo      = (const float*)d_in[12];
    const float* damp_min = (const float*)d_in[13];
    float* out = (float*)d_out;

    half_t* ws = (half_t*)d_ws;

    // Pre-pass: pack all weights to f16 in WMMA-B fragment order.
    auto pack = [&](const float* W, size_t off, int K, int N) {
        int total = K * N;
        pack_weight_f16<<<(total + 255) / 256, 256, 0, stream>>>(W, ws + off, K, N);
    };
    pack(Wd1, WS_WD1, NN, HH);
    pack(Wd2, WS_WD2, HH, HH);
    pack(Wdo, WS_WDO, HH, NN);
    pack(Wo1, WS_WO1, NN, HH);
    pack(Wo2, WS_WO2, HH, HH);
    pack(Woo, WS_WOO, HH, OFFN);

    const int Brows = in_sizes[0] / NN;          // 32768
    const int nWG   = Brows / 16;                // 2048 workgroups
    const size_t smem = (size_t)(3 * 16 * NN + 16 * OFFN) * sizeof(float)
                      + (size_t)(16 * NN + 2 * 16 * HH) * sizeof(half_t); // 159744 B

    damping_fused_kernel<<<nWG, 256, smem, stream>>>(
        x, bd1, bd2, bdo, bo1, bo2, boo, damp_min, ws, out);
}
